// MultiHeadMaskedAttention_3530463117337
// MI455X (gfx1250) — matrix operand order repaired and checked by reading, not compiled, not run on hardware
//
#include <hip/hip_runtime.h>
#include <hip/hip_bf16.h>

typedef __attribute__((ext_vector_type(16))) __bf16 v16bf;
typedef __attribute__((ext_vector_type(8)))  __bf16 v8bf;
typedef __attribute__((ext_vector_type(8)))  float  v8f;

#define WMMA_BF16(a, b, c) \
  __builtin_amdgcn_wmma_f32_16x16x32_bf16(false, (a), false, (b), (short)0, (c), false, false)

static constexpr int BB = 4;      // batch
static constexpr int S  = 2048;   // sequence
static constexpr int E  = 1024;   // embedding
static constexpr int H  = 16;     // heads
static constexpr int D  = 64;     // head dim
static constexpr float SCALE = 0.03125f;  // 1/sqrt(E) = 1/32

// Build a 16-element bf16 fragment vector from two 8-element (16-byte) runs.
__device__ __forceinline__ v16bf ldpair(const __bf16* p0, const __bf16* p1) {
  v8bf a = *(const v8bf*)p0;
  v8bf b = *(const v8bf*)p1;
  v16bf r;
  #pragma unroll
  for (int j = 0; j < 8; ++j) { r[j] = a[j]; r[j + 8] = b[j]; }
  return r;
}

// ---------------------------------------------------------------------------
// Kernel 0: fp32 -> bf16 bulk convert (8 elements per thread, b128 in/out)
// ---------------------------------------------------------------------------
__global__ __launch_bounds__(256)
void f32_to_bf16_kernel(const float* __restrict__ in, __bf16* __restrict__ outp, int n8) {
  const int t = blockIdx.x * 256 + threadIdx.x;
  if (t >= n8) return;
  const float4* p = (const float4*)(in + (size_t)t * 8);
  float4 f0 = p[0], f1 = p[1];
  v8bf r;
  r[0] = (__bf16)f0.x; r[1] = (__bf16)f0.y; r[2] = (__bf16)f0.z; r[3] = (__bf16)f0.w;
  r[4] = (__bf16)f1.x; r[5] = (__bf16)f1.y; r[6] = (__bf16)f1.z; r[7] = (__bf16)f1.w;
  *(v8bf*)(outp + (size_t)t * 8) = r;
}

// ---------------------------------------------------------------------------
// Kernel 1: Q = x @ Wq^T, K = x @ Wk^T (bf16 in, f32 acc, bf16 out head-split
// [B,H,S,D]). One wave computes a 64x16 tile of BOTH Q and K:
// 4 A-fragments (shared by Q and K) x 1 B-fragment each -> 8 WMMA / k-step.
// ---------------------------------------------------------------------------
__global__ __launch_bounds__(256)
void qk_proj_kernel(const __bf16* __restrict__ x,  const __bf16* __restrict__ Wq,
                    const __bf16* __restrict__ Wk, __bf16* __restrict__ Qbuf,
                    __bf16* __restrict__ Kbuf) {
  const int lane = threadIdx.x & 31;
  const int wave = threadIdx.x >> 5;
  const int w    = blockIdx.x * 8 + wave;
  const int mt = w >> 6, nt = w & 63;      // Ntiles = E/16 = 64
  const int m0 = mt * 64, n0 = nt * 16;
  const int lo = lane & 15, hi = lane >> 4;

  const __bf16* xr0 = x + (size_t)(m0 + lo)      * E + hi * 8;
  const __bf16* xr1 = x + (size_t)(m0 + 16 + lo) * E + hi * 8;
  const __bf16* xr2 = x + (size_t)(m0 + 32 + lo) * E + hi * 8;
  const __bf16* xr3 = x + (size_t)(m0 + 48 + lo) * E + hi * 8;
  const __bf16* wq  = Wq + (size_t)(n0 + lo) * E + hi * 8;
  const __bf16* wk  = Wk + (size_t)(n0 + lo) * E + hi * 8;

  v8f cq0 = {}, cq1 = {}, cq2 = {}, cq3 = {};
  v8f ck0 = {}, ck1 = {}, ck2 = {}, ck3 = {};
  for (int k0 = 0; k0 < E; k0 += 32) {
    v16bf bq = ldpair(wq + k0, wq + k0 + 16);
    v16bf bk = ldpair(wk + k0, wk + k0 + 16);
    v16bf a0 = ldpair(xr0 + k0, xr0 + k0 + 16);
    v16bf a1 = ldpair(xr1 + k0, xr1 + k0 + 16);
    v16bf a2 = ldpair(xr2 + k0, xr2 + k0 + 16);
    v16bf a3 = ldpair(xr3 + k0, xr3 + k0 + 16);
    cq0 = WMMA_BF16(a0, bq, cq0);  ck0 = WMMA_BF16(a0, bk, ck0);
    cq1 = WMMA_BF16(a1, bq, cq1);  ck1 = WMMA_BF16(a1, bk, ck1);
    cq2 = WMMA_BF16(a2, bq, cq2);  ck2 = WMMA_BF16(a2, bk, ck2);
    cq3 = WMMA_BF16(a3, bq, cq3);  ck3 = WMMA_BF16(a3, bk, ck3);
  }

  const int col = n0 + lo;
  const int h = col >> 6, d = col & 63;
  v8f* cqs[4] = { &cq0, &cq1, &cq2, &cq3 };
  v8f* cks[4] = { &ck0, &ck1, &ck2, &ck3 };
  #pragma unroll
  for (int r = 0; r < 4; ++r) {
    #pragma unroll
    for (int i = 0; i < 8; ++i) {
      const int m = m0 + 16 * r + i + 8 * hi;
      const int b = m >> 11, s = m & (S - 1);
      const size_t idx = (((size_t)b * H + h) * S + s) * D + d;
      Qbuf[idx] = (__bf16)((*cqs[r])[i]);
      Kbuf[idx] = (__bf16)((*cks[r])[i]);
    }
  }
}

// ---------------------------------------------------------------------------
// Kernel 2: causal flash attention, 32-row q tile per wave (two 16-row
// halves share every K-fragment load and V gather); V = Q (ref bug).
// Full key-blocks run a compare-free softmax; exactly one diagonal block
// (kb == q0) uses the masked path.
// ---------------------------------------------------------------------------
template <bool MASKED>
__device__ __forceinline__ void softmax_half(
    const v8f& s0, const v8f& s1, int kb, int qbase, int lo, int hi,
    float* mrow, float* lrow, v8f* o, __bf16* plhalf) {
  #pragma unroll
  for (int i = 0; i < 8; ++i) {
    float v0 = s0[i] * SCALE;
    float v1 = s1[i] * SCALE;
    if (MASKED) {
      const int qrow = qbase + i + 8 * hi;
      if (kb + lo      > qrow) v0 = -1.0e30f;
      if (kb + 16 + lo > qrow) v1 = -1.0e30f;
    }
    float mx = fmaxf(v0, v1);
    #pragma unroll
    for (int msk = 1; msk < 16; msk <<= 1) mx = fmaxf(mx, __shfl_xor(mx, msk, 32));
    const float mnew  = fmaxf(mrow[i], mx);
    const float alpha = __expf(mrow[i] - mnew);
    const float p0 = __expf(v0 - mnew);
    const float p1 = __expf(v1 - mnew);
    float sum = p0 + p1;
    #pragma unroll
    for (int msk = 1; msk < 16; msk <<= 1) sum += __shfl_xor(sum, msk, 32);
    lrow[i] = lrow[i] * alpha + sum;
    mrow[i] = mnew;
    o[0][i] *= alpha; o[1][i] *= alpha; o[2][i] *= alpha; o[3][i] *= alpha;
    const int r = i + 8 * hi;             // P tile (C layout) -> LDS row-major
    plhalf[r * 32 + lo]      = (__bf16)p0;
    plhalf[r * 32 + 16 + lo] = (__bf16)p1;
  }
}

template <bool MASKED>
__device__ __forceinline__ void attn_block32(
    int kb, int q0, int lo, int hi,
    const __bf16* __restrict__ Qh, const __bf16* __restrict__ Kh,
    const v16bf& aq00, const v16bf& aq01,     // q-half A: rows q0..q0+15, d 0..31 / 32..63
    const v16bf& aq10, const v16bf& aq11,     // q-half B: rows q0+16..q0+31
    v8f* oA, v8f* oB,
    float* mrowA, float* lrowA, float* mrowB, float* lrowB,
    __bf16* pl) {
  // ---- shared K fragments: keys kb..kb+15 and kb+16..kb+31, d-chunks 0/1
  const __bf16* kp0 = Kh + (size_t)(kb + lo)      * D + hi * 8;
  const __bf16* kp1 = Kh + (size_t)(kb + 16 + lo) * D + hi * 8;
  const v16bf b0 = ldpair(kp0,      kp0 + 16);
  const v16bf b1 = ldpair(kp1,      kp1 + 16);
  const v16bf c0 = ldpair(kp0 + 32, kp0 + 48);
  const v16bf c1 = ldpair(kp1 + 32, kp1 + 48);

  v8f sA0 = {}, sA1 = {}, sB0 = {}, sB1 = {};
  sA0 = WMMA_BF16(aq00, b0, sA0);  sA1 = WMMA_BF16(aq00, b1, sA1);
  sB0 = WMMA_BF16(aq10, b0, sB0);  sB1 = WMMA_BF16(aq10, b1, sB1);
  sA0 = WMMA_BF16(aq01, c0, sA0);  sA1 = WMMA_BF16(aq01, c1, sA1);
  sB0 = WMMA_BF16(aq11, c0, sB0);  sB1 = WMMA_BF16(aq11, c1, sB1);

  softmax_half<MASKED>(sA0, sA1, kb, q0,      lo, hi, mrowA, lrowA, oA, pl);
  softmax_half<MASKED>(sB0, sB1, kb, q0 + 16, lo, hi, mrowB, lrowB, oB, pl + 16 * 32);

  // ---- P A-fragments (transpose via LDS), one per q-half
  v16bf ap0, ap1;
  {
    const __bf16* base0 = pl + lo * 32 + hi * 16;
    const __bf16* base1 = base0 + 16 * 32;
    v8bf x0 = *(const v8bf*)(base0);
    v8bf x1 = *(const v8bf*)(base0 + 8);
    v8bf y0 = *(const v8bf*)(base1);
    v8bf y1 = *(const v8bf*)(base1 + 8);
    #pragma unroll
    for (int j = 0; j < 8; ++j) {
      ap0[j] = x0[j]; ap0[j + 8] = x1[j];
      ap1[j] = y0[j]; ap1[j + 8] = y1[j];
    }
  }

  // ---- O += P @ V (V = Q); each gathered V fragment feeds BOTH q-halves
  const __bf16* vbase = Qh + (size_t)(kb + hi * 16) * D + lo;
  v16bf bv;
  #pragma unroll
  for (int j = 0; j < 16; ++j) bv[j] = vbase[(size_t)j * D];
  oA[0] = WMMA_BF16(ap0, bv, oA[0]);  oB[0] = WMMA_BF16(ap1, bv, oB[0]);
  #pragma unroll
  for (int j = 0; j < 16; ++j) bv[j] = vbase[(size_t)j * D + 16];
  oA[1] = WMMA_BF16(ap0, bv, oA[1]);  oB[1] = WMMA_BF16(ap1, bv, oB[1]);
  #pragma unroll
  for (int j = 0; j < 16; ++j) bv[j] = vbase[(size_t)j * D + 32];
  oA[2] = WMMA_BF16(ap0, bv, oA[2]);  oB[2] = WMMA_BF16(ap1, bv, oB[2]);
  #pragma unroll
  for (int j = 0; j < 16; ++j) bv[j] = vbase[(size_t)j * D + 48];
  oA[3] = WMMA_BF16(ap0, bv, oA[3]);  oB[3] = WMMA_BF16(ap1, bv, oB[3]);
}

__global__ __launch_bounds__(256)
void attn_kernel(const __bf16* __restrict__ Qbuf, const __bf16* __restrict__ Kbuf,
                 __bf16* __restrict__ Abuf) {
  __shared__ alignas(16) __bf16 Plds[8][32 * 32];   // per-wave 32x32 bf16 P tile

  const int lane = threadIdx.x & 31;
  const int wid  = threadIdx.x >> 5;
  const int w    = blockIdx.x * 8 + wid;
  const int QT   = S / 32;                  // 64 q-tiles per (b,h)
  const int b    = w / (H * QT);
  const int rem  = w % (H * QT);
  const int h    = rem / QT;
  const int q0   = (rem % QT) * 32;

  const __bf16* Qh = Qbuf + ((size_t)b * H + h) * (size_t)S * D;
  const __bf16* Kh = Kbuf + ((size_t)b * H + h) * (size_t)S * D;
  const int lo = lane & 15, hi = lane >> 4;

  const __bf16* qrA = Qh + (size_t)(q0 + lo)      * D + hi * 8;
  const __bf16* qrB = Qh + (size_t)(q0 + 16 + lo) * D + hi * 8;
  const v16bf aq00 = ldpair(qrA,      qrA + 16);
  const v16bf aq01 = ldpair(qrA + 32, qrA + 48);
  const v16bf aq10 = ldpair(qrB,      qrB + 16);
  const v16bf aq11 = ldpair(qrB + 32, qrB + 48);

  v8f oA[4] = {{},{},{},{}};
  v8f oB[4] = {{},{},{},{}};
  float mrowA[8], lrowA[8], mrowB[8], lrowB[8];
  #pragma unroll
  for (int i = 0; i < 8; ++i) {
    mrowA[i] = -INFINITY; lrowA[i] = 0.0f;
    mrowB[i] = -INFINITY; lrowB[i] = 0.0f;
  }

  __bf16* pl = Plds[wid];

  for (int kb = 0; kb < q0; kb += 32)       // full (mask-free) key blocks
    attn_block32<false>(kb, q0, lo, hi, Qh, Kh, aq00, aq01, aq10, aq11,
                        oA, oB, mrowA, lrowA, mrowB, lrowB, pl);
  attn_block32<true>(q0, q0, lo, hi, Qh, Kh, aq00, aq01, aq10, aq11,
                     oA, oB, mrowA, lrowA, mrowB, lrowB, pl);  // diagonal block

  // ---- finalize: divide by l, write attn merged-head [B,S,E] bf16
  #pragma unroll
  for (int i = 0; i < 8; ++i) {
    {
      const int   srow = q0 + i + 8 * hi;
      const float inv  = 1.0f / lrowA[i];
      __bf16* orow = Abuf + ((size_t)b * S + srow) * E + h * D + lo;
      orow[0]  = (__bf16)(oA[0][i] * inv);
      orow[16] = (__bf16)(oA[1][i] * inv);
      orow[32] = (__bf16)(oA[2][i] * inv);
      orow[48] = (__bf16)(oA[3][i] * inv);
    }
    {
      const int   srow = q0 + 16 + i + 8 * hi;
      const float inv  = 1.0f / lrowB[i];
      __bf16* orow = Abuf + ((size_t)b * S + srow) * E + h * D + lo;
      orow[0]  = (__bf16)(oB[0][i] * inv);
      orow[16] = (__bf16)(oB[1][i] * inv);
      orow[32] = (__bf16)(oB[2][i] * inv);
      orow[48] = (__bf16)(oB[3][i] * inv);
    }
  }
}

// ---------------------------------------------------------------------------
// Kernel 3: out = attn @ Wo^T + bo (bf16 in, f32 out). 64x16 tile per wave.
// ---------------------------------------------------------------------------
__global__ __launch_bounds__(256)
void out_proj_kernel(const __bf16* __restrict__ A, const __bf16* __restrict__ Wo,
                     const float* __restrict__ bo, float* __restrict__ out) {
  const int lane = threadIdx.x & 31;
  const int wave = threadIdx.x >> 5;
  const int w    = blockIdx.x * 8 + wave;
  const int mt = w >> 6, nt = w & 63;
  const int m0 = mt * 64, n0 = nt * 16;
  const int lo = lane & 15, hi = lane >> 4;

  const __bf16* ar0 = A + (size_t)(m0 + lo)      * E + hi * 8;
  const __bf16* ar1 = A + (size_t)(m0 + 16 + lo) * E + hi * 8;
  const __bf16* ar2 = A + (size_t)(m0 + 32 + lo) * E + hi * 8;
  const __bf16* ar3 = A + (size_t)(m0 + 48 + lo) * E + hi * 8;
  const __bf16* wr  = Wo + (size_t)(n0 + lo) * E + hi * 8;

  v8f c0 = {}, c1 = {}, c2 = {}, c3 = {};
  for (int k0 = 0; k0 < E; k0 += 32) {
    v16bf bw = ldpair(wr + k0, wr + k0 + 16);
    v16bf a0 = ldpair(ar0 + k0, ar0 + k0 + 16);
    v16bf a1 = ldpair(ar1 + k0, ar1 + k0 + 16);
    v16bf a2 = ldpair(ar2 + k0, ar2 + k0 + 16);
    v16bf a3 = ldpair(ar3 + k0, ar3 + k0 + 16);
    c0 = WMMA_BF16(a0, bw, c0);
    c1 = WMMA_BF16(a1, bw, c1);
    c2 = WMMA_BF16(a2, bw, c2);
    c3 = WMMA_BF16(a3, bw, c3);
  }

  const int   col  = n0 + lo;
  const float bias = bo[col];
  v8f* cs[4] = { &c0, &c1, &c2, &c3 };
  #pragma unroll
  for (int r = 0; r < 4; ++r) {
    #pragma unroll
    for (int i = 0; i < 8; ++i) {
      const int m = m0 + 16 * r + i + 8 * hi;
      out[(size_t)m * E + col] = (*cs[r])[i] + bias;
    }
  }
}

// ---------------------------------------------------------------------------
extern "C" void kernel_launch(void* const* d_in, const int* in_sizes, int n_in,
                              void* d_out, int out_size, void* d_ws, size_t ws_size,
                              hipStream_t stream) {
  const float* x  = (const float*)d_in[0];
  const float* Wk = (const float*)d_in[1];
  const float* Wq = (const float*)d_in[2];
  // d_in[3] (Wv) intentionally unused: reference discards x@Wv^T (V = Q bug).
  const float* Wo = (const float*)d_in[4];
  const float* bo = (const float*)d_in[5];
  float* out = (float*)d_out;

  const size_t NX = (size_t)BB * S * E;     // 8,388,608
  const size_t NW = (size_t)E * E;          // 1,048,576
  char* ws = (char*)d_ws;
  __bf16* xbf  = (__bf16*)(ws);                                   // 16 MB
  __bf16* Wqbf = (__bf16*)(ws + 2 * NX);                          //  2 MB
  __bf16* Wkbf = (__bf16*)(ws + 2 * NX + 2 * NW);                 //  2 MB
  __bf16* Wobf = (__bf16*)(ws + 2 * NX + 4 * NW);                 //  2 MB
  __bf16* Qbf  = (__bf16*)(ws + 2 * NX + 6 * NW);                 // 16 MB
  __bf16* Kbf  = (__bf16*)(ws + 4 * NX + 6 * NW);                 // 16 MB
  __bf16* Abf  = (__bf16*)(ws + 6 * NX + 6 * NW);                 // 16 MB

  f32_to_bf16_kernel<<<(int)(NX / 8 / 256), 256, 0, stream>>>(x,  xbf,  (int)(NX / 8));
  f32_to_bf16_kernel<<<(int)(NW / 8 / 256), 256, 0, stream>>>(Wq, Wqbf, (int)(NW / 8));
  f32_to_bf16_kernel<<<(int)(NW / 8 / 256), 256, 0, stream>>>(Wk, Wkbf, (int)(NW / 8));
  f32_to_bf16_kernel<<<(int)(NW / 8 / 256), 256, 0, stream>>>(Wo, Wobf, (int)(NW / 8));

  // waves: proj/out-proj = (B*S/64)*(E/16) = 8192 ; attn = B*H*(S/32) = 4096
  qk_proj_kernel <<<1024, 256, 0, stream>>>(xbf, Wqbf, Wkbf, Qbf, Kbf);
  attn_kernel    <<< 512, 256, 0, stream>>>(Qbf, Kbf, Abf);
  out_proj_kernel<<<1024, 256, 0, stream>>>(Abf, Wobf, bo, out);
}
